// PointNet2SemSegMsg_41274635714775
// MI455X (gfx1250) — matrix operand order repaired and checked by reading, not compiled, not run on hardware
//
#include <hip/hip_runtime.h>
#include <cstdint>
#include <cstddef>

// ---------------------------------------------------------------------------
// Types for CDNA5 WMMA (wave32): v_wmma_f32_16x16x32_bf16
// ---------------------------------------------------------------------------
typedef __bf16 bf16_t;
typedef __attribute__((ext_vector_type(16))) __bf16 v16bf;
typedef __attribute__((ext_vector_type(8)))  __bf16 v8bf;
typedef __attribute__((ext_vector_type(8)))  float  v8f;
typedef __attribute__((ext_vector_type(4)))  int    v4i_t;

#define BN_SCALE_C 0.999995000037f  // 1/sqrt(1+1e-5)

// gfx1250 async global->LDS path (GLOBAL_LOAD_ASYNC_TO_LDS_B128, ASYNCcnt)
#if defined(__has_builtin)
#  if __has_builtin(__builtin_amdgcn_global_load_async_to_lds_b128) && \
      __has_builtin(__builtin_amdgcn_s_wait_asynccnt)
#    define HAVE_ASYNC_LDS 1
#  endif
#endif
#ifndef HAVE_ASYNC_LDS
#  define HAVE_ASYNC_LDS 0
#endif

// Typed address-space pointer casts for the async builtin:
//   param0: int4 __device__ * (AS1 global), param1: int4 LDS (AS3)
#define ASYNC_GPTR(p) ((__attribute__((address_space(1))) v4i_t*)(p))
#define ASYNC_LPTR(p) ((__attribute__((address_space(3))) v4i_t*)(p))

__device__ __forceinline__ bf16_t f2bf(float f) {
  union { float f; uint32_t u; } v; v.f = f;
  uint32_t r = (v.u + 0x7FFFu + ((v.u >> 16) & 1u)) >> 16;  // RNE
  union { uint16_t s; bf16_t b; } o; o.s = (uint16_t)r; return o.b;
}
__device__ __forceinline__ float bf2f(bf16_t b) {
  union { uint16_t s; bf16_t b; } i; i.b = b;
  union { uint32_t u; float f; } o; o.u = ((uint32_t)i.s) << 16; return o.f;
}

// ---------------------------------------------------------------------------
// Stage one 64(N) x 64(K) bf16 weight chunk into LDS (8 KB), 256 threads,
// 2 x 16B per thread. Async-to-LDS when available, else load+ds_store.
// ---------------------------------------------------------------------------
__device__ __forceinline__ void stage_b64(bf16_t* __restrict__ ldst,
                                          const bf16_t* __restrict__ gsrc,
                                          int Kp, int tid)
{
#pragma unroll
  for (int t = 0; t < 2; ++t) {
    int chunk = tid + t * 256;            // 512 chunks of 16B
    int row = chunk >> 3;                 // 8 chunks per 128B row
    int ko  = (chunk & 7) << 3;           // bf16 offset within row
    const bf16_t* g = gsrc + (size_t)row * Kp + ko;
    bf16_t* l = ldst + row * 64 + ko;
#if HAVE_ASYNC_LDS
    __builtin_amdgcn_global_load_async_to_lds_b128(ASYNC_GPTR(g), ASYNC_LPTR(l), 0, 0);
#else
    *(uint4*)l = *(const uint4*)g;        // global_load_b128 + ds_store_b128
#endif
  }
}

// ---------------------------------------------------------------------------
// Register-blocked WMMA GEMM:  Y[M,Np] = epilogue( X[M,Kp] @ Wp[Np,Kp]^T )
//   - block = 8 waves = 128-row M stripe; 64-col N stripe
//   - each wave: one 16-row tile x 4 N-tiles (4 accumulators, A reused 4x)
//   - W stripe double-buffered through LDS in 64x64 chunks (async-to-LDS)
//   - Kp, Np multiples of 64; M multiple of 128 (no partial blocks -> EXEC=~0)
//   epilogue: relu(g * (acc + b) * BN_SCALE + be)  (or linear for logits)
// ---------------------------------------------------------------------------
__global__ void __launch_bounds__(256) wmma_gemm_tiled(
    const bf16_t* __restrict__ X, const bf16_t* __restrict__ Wp,
    const float* __restrict__ ep, bf16_t* __restrict__ Ybf,
    float* __restrict__ Yf, int M, int Np, int Kp, int ldy, int bnrelu)
{
  __shared__ __align__(32) bf16_t sB[2][64 * 64];   // 16 KB double buffer
  const int nstripes = Np >> 6;
  int ms = blockIdx.x / nstripes;
  int ns = blockIdx.x - ms * nstripes;
  int tid = threadIdx.x;
  int wv = tid >> 5, lane = tid & 31, half = lane >> 4, r16 = lane & 15;

  // A: row (ms*128 + wv*16 + r16); K chunks at k+half*8 / k+16+half*8 (ISA 7.12.2)
  const bf16_t* xrow  = X  + (size_t)(ms * 128 + wv * 16 + r16) * Kp + half * 8;
  const bf16_t* wbase = Wp + (size_t)(ns * 64) * Kp;

  v8f zero = {0.f, 0.f, 0.f, 0.f, 0.f, 0.f, 0.f, 0.f};
  v8f acc[4];
#pragma unroll
  for (int t = 0; t < 4; ++t) acc[t] = zero;

  stage_b64(&sB[0][0], wbase, Kp, tid);
#if HAVE_ASYNC_LDS
  __builtin_amdgcn_s_wait_asynccnt(0);
#endif
  __syncthreads();

  int nk = Kp >> 6;
  for (int c = 0; c < nk; ++c) {
    int buf = c & 1;
    if (c + 1 < nk)
      stage_b64(&sB[buf ^ 1][0], wbase + (size_t)(c + 1) * 64, Kp, tid);

    const bf16_t* sb = &sB[buf][0];
#pragma unroll
    for (int kk = 0; kk < 64; kk += 32) {
      int k = (c << 6) + kk;
      union { v16bf v; v8bf h[2]; } a;
      a.h[0] = *(const v8bf*)(xrow + k);
      a.h[1] = *(const v8bf*)(xrow + k + 16);
      __builtin_prefetch(xrow + k + 256, 0, 3);   // global_prefetch_b8
#pragma unroll
      for (int t = 0; t < 4; ++t) {
        // B column tile t: row (t*16+r16) of W stripe, K chunks at kk+half*8 / kk+16+half*8
        union { v16bf v; v8bf h[2]; } bu;
        bu.h[0] = *(const v8bf*)(sb + (t * 16 + r16) * 64 + kk + half * 8);
        bu.h[1] = *(const v8bf*)(sb + (t * 16 + r16) * 64 + kk + 16 + half * 8);
        v16bf bm = bu.v;
        acc[t] = __builtin_amdgcn_wmma_f32_16x16x32_bf16(
            false, a.v, false, bm, (short)0, acc[t], false, false);
      }
    }
#if HAVE_ASYNC_LDS
    __builtin_amdgcn_s_wait_asynccnt(0);
#endif
    __syncthreads();
  }

#pragma unroll
  for (int t = 0; t < 4; ++t) {
    int n = ns * 64 + t * 16 + r16;
    float bias = ep[n], gg = ep[Np + n], be = ep[2 * Np + n];
#pragma unroll
    for (int r = 0; r < 8; ++r) {
      int m = ms * 128 + wv * 16 + half * 8 + r;   // D VGPR r <-> row half*8+r
      float v = acc[t][r] + bias;
      if (bnrelu) v = fmaxf(gg * v * BN_SCALE_C + be, 0.0f);
      if (Ybf) Ybf[(size_t)m * ldy + n] = f2bf(v);
      if (Yf)  Yf [(size_t)m * ldy + n] = v;
    }
  }
}

// ---------------------------------------------------------------------------
// Stage a layer's weights: f32 [cout,cin] -> bf16 [Np,Kp] zero-padded, plus
// padded epilogue params ep = [bias|gamma|beta].
// ---------------------------------------------------------------------------
__global__ void prep_weights(const float* __restrict__ W, const float* __restrict__ bias,
                             const float* __restrict__ g, const float* __restrict__ be,
                             bf16_t* __restrict__ Wp, float* __restrict__ ep,
                             int cout, int cin, int Np, int Kp)
{
  int gid = blockIdx.x * blockDim.x + threadIdx.x;
  int tot = Np * Kp;
  if (gid < tot) {
    int n = gid / Kp, k = gid - n * Kp;
    Wp[gid] = (n < cout && k < cin) ? f2bf(W[(size_t)n * cin + k]) : f2bf(0.0f);
  }
  if (gid < Np) {
    bool v = gid < cout;
    ep[gid]          = v ? bias[gid] : 0.0f;
    ep[Np + gid]     = v ? g[gid]    : 0.0f;
    ep[2 * Np + gid] = v ? be[gid]   : 0.0f;
  }
}

// ---------------------------------------------------------------------------
// coord (B,9,Npt) -> feat0 bf16 (B,Npt,ldf) + xyz f32 (B,Npt,3)
// ---------------------------------------------------------------------------
__global__ void prep_coord(const float* __restrict__ coord, float* __restrict__ xyz,
                           bf16_t* __restrict__ feat, int B, int Npt, int Cin, int ldf)
{
  int gid = blockIdx.x * blockDim.x + threadIdx.x;
  if (gid >= B * Npt) return;
  int b = gid / Npt, n = gid - b * Npt;
  for (int c = 0; c < ldf; ++c) {
    float v = (c < Cin) ? coord[((size_t)b * Cin + c) * Npt + n] : 0.0f;
    feat[(size_t)gid * ldf + c] = f2bf(v);
    if (c < 3) xyz[(size_t)gid * 3 + c] = v;
  }
}

// ---------------------------------------------------------------------------
// Farthest point sampling: one block per batch, LDS argmax reduction.
// Emits the index BEFORE the update step (matches reference scan).
// ---------------------------------------------------------------------------
__global__ void __launch_bounds__(256) fps_kernel(
    const float* __restrict__ xyz, int N, int S,
    int* __restrict__ outidx, float* __restrict__ new_xyz)
{
  const int T = 256;
  int b = blockIdx.x, tid = threadIdx.x;
  __shared__ float dist[4096];
  __shared__ float rmax[T];
  __shared__ int   rarg[T];
  __shared__ int   sfar;
  const float* xb = xyz + (size_t)b * N * 3;
  for (int i = tid; i < N; i += T) dist[i] = 1e10f;
  if (tid == 0) sfar = 0;
  __syncthreads();
  for (int s = 0; s < S; ++s) {
    int far = sfar;
    float cx = xb[far * 3 + 0], cy = xb[far * 3 + 1], cz = xb[far * 3 + 2];
    if (tid == 0) {
      outidx[b * S + s] = far;
      float* nz = new_xyz + ((size_t)b * S + s) * 3;
      nz[0] = cx; nz[1] = cy; nz[2] = cz;
    }
    float lm = -1.0f; int la = 0;
    for (int i = tid; i < N; i += T) {
      float dx = xb[i * 3] - cx, dy = xb[i * 3 + 1] - cy, dz = xb[i * 3 + 2] - cz;
      float nd = fminf(dist[i], dx * dx + dy * dy + dz * dz);
      dist[i] = nd;
      if (nd > lm) { lm = nd; la = i; }
    }
    rmax[tid] = lm; rarg[tid] = la;
    __syncthreads();
    for (int st = T >> 1; st > 0; st >>= 1) {
      if (tid < st && rmax[tid + st] > rmax[tid]) {
        rmax[tid] = rmax[tid + st]; rarg[tid] = rarg[tid + st];
      }
      __syncthreads();
    }
    if (tid == 0) sfar = rarg[0];
    __syncthreads();
  }
}

// ---------------------------------------------------------------------------
// Ball query (first K in-radius indices, pad with first) + grouped gather:
// X row = [ feat[g] (Cf ch) | xyz[g]-new_xyz (3 ch) | zero pad to ldx ]
// ---------------------------------------------------------------------------
__global__ void __launch_bounds__(64) ball_gather(
    const float* __restrict__ xyz, const float* __restrict__ new_xyz,
    const bf16_t* __restrict__ feat, int Cf, int ldf,
    bf16_t* __restrict__ X, int N, int S, int K, int ldx, float r2)
{
  int b = blockIdx.x / S, s = blockIdx.x - b * S;
  int tid = threadIdx.x;
  __shared__ int sidx[32];
  const float* xb = xyz + (size_t)b * N * 3;
  const float* q  = new_xyz + ((size_t)b * S + s) * 3;
  float qx = q[0], qy = q[1], qz = q[2];
  if (tid == 0) {
    int c = 0;
    for (int j = 0; j < N && c < K; ++j) {
      float dx = xb[j * 3] - qx, dy = xb[j * 3 + 1] - qy, dz = xb[j * 3 + 2] - qz;
      if (dx * dx + dy * dy + dz * dz <= r2) sidx[c++] = j;
    }
    if (c == 0) sidx[c++] = 0;
    int f = sidx[0];
    for (int t = c; t < K; ++t) sidx[t] = f;
  }
  __syncthreads();
  for (int kk = 0; kk < K; ++kk) {
    int g = sidx[kk];
    size_t row = (((size_t)b * S + s) * K + kk) * (size_t)ldx;
    for (int c = tid; c < ldx; c += 64) {
      float v;
      if (c < Cf)           v = bf2f(feat[((size_t)b * N + g) * ldf + c]);
      else if (c < Cf + 3)  v = xb[g * 3 + (c - Cf)] - ((c - Cf) == 0 ? qx : (c - Cf) == 1 ? qy : qz);
      else                  v = 0.0f;
      X[row + c] = f2bf(v);
    }
  }
}

// ---------------------------------------------------------------------------
// Max over the K group dimension -> write into concat feature buffer at coff.
// ---------------------------------------------------------------------------
__global__ void maxpool_kernel(const bf16_t* __restrict__ Xin, bf16_t* __restrict__ feat_out,
                               int K, int C, int ld, int ldo, int coff)
{
  int bs = blockIdx.x, tid = threadIdx.x;
  for (int c = tid; c < C; c += blockDim.x) {
    float m = -1e30f;
    for (int kk = 0; kk < K; ++kk)
      m = fmaxf(m, bf2f(Xin[((size_t)bs * K + kk) * ld + c]));
    feat_out[(size_t)bs * ldo + coff + c] = f2bf(m);
  }
}

// ---------------------------------------------------------------------------
// Feature propagation gather: 3-NN inverse-distance interpolation of feat2,
// concat with feat1, zero-pad to ldx. One thread per (b, n1).
// ---------------------------------------------------------------------------
__global__ void fp_gather(const float* __restrict__ xyz1, const float* __restrict__ xyz2,
                          const bf16_t* __restrict__ f1, int C1, int ld1,
                          const bf16_t* __restrict__ f2, int C2, int ld2,
                          bf16_t* __restrict__ X, int N1, int N2, int ldx, int total)
{
  int gid = blockIdx.x * blockDim.x + threadIdx.x;
  if (gid >= total) return;
  int b = gid / N1, n = gid - b * N1;
  const float* p = xyz1 + ((size_t)b * N1 + n) * 3;
  float px = p[0], py = p[1], pz = p[2];
  const float* q = xyz2 + (size_t)b * N2 * 3;
  float d0 = 1e30f, d1 = 1e30f, d2v = 1e30f; int i0 = 0, i1 = 0, i2 = 0;
  for (int j = 0; j < N2; ++j) {
    float dx = q[j * 3] - px, dy = q[j * 3 + 1] - py, dz = q[j * 3 + 2] - pz;
    float d = dx * dx + dy * dy + dz * dz;
    if (d < d0)       { d2v = d1; i2 = i1; d1 = d0; i1 = i0; d0 = d; i0 = j; }
    else if (d < d1)  { d2v = d1; i2 = i1; d1 = d; i1 = j; }
    else if (d < d2v) { d2v = d; i2 = j; }
  }
  float w0 = 1.0f / (d0 + 1e-8f), w1 = 1.0f / (d1 + 1e-8f), w2 = 1.0f / (d2v + 1e-8f);
  float ws = w0 + w1 + w2; w0 /= ws; w1 /= ws; w2 /= ws;
  size_t row = (size_t)gid * ldx;
  for (int c = 0; c < C1; ++c)
    X[row + c] = f1[((size_t)b * N1 + n) * ld1 + c];
  for (int c = 0; c < C2; ++c) {
    float v = w0 * bf2f(f2[((size_t)b * N2 + i0) * ld2 + c])
            + w1 * bf2f(f2[((size_t)b * N2 + i1) * ld2 + c])
            + w2 * bf2f(f2[((size_t)b * N2 + i2) * ld2 + c]);
    X[row + C1 + c] = f2bf(v);
  }
  for (int c = C1 + C2; c < ldx; ++c) X[row + c] = f2bf(0.0f);
}

// ---------------------------------------------------------------------------
// log_softmax over the 13 valid logits per point -> d_out.
// ---------------------------------------------------------------------------
__global__ void logsoftmax_kernel(const float* __restrict__ logits, int ld, int C,
                                  float* __restrict__ out, int rows)
{
  int gid = blockIdx.x * blockDim.x + threadIdx.x;
  if (gid >= rows) return;
  const float* x = logits + (size_t)gid * ld;
  float m = -1e30f;
  for (int c = 0; c < C; ++c) m = fmaxf(m, x[c]);
  float s = 0.0f;
  for (int c = 0; c < C; ++c) s += expf(x[c] - m);
  float ls = logf(s);
  for (int c = 0; c < C; ++c) out[(size_t)gid * C + c] = x[c] - m - ls;
}

// new_xyz4 (16,16,3) -> (16,3,16)
__global__ void newxyz_kernel(const float* __restrict__ nx, float* __restrict__ out)
{
  int gid = blockIdx.x * blockDim.x + threadIdx.x;
  if (gid >= 16 * 3 * 16) return;
  int b = gid / 48, rem = gid - b * 48, c = rem / 16, s = rem - c * 16;
  out[gid] = nx[((size_t)b * 16 + s) * 3 + c];
}

// ---------------------------------------------------------------------------
// Host orchestration
// Param flattening (JAX sorted-key order): coord, fp1(3L), fp2(2L), fp3(2L),
// fp4(2L), head_W, head_b, head_mlp(1L), sa1(2x3L), sa2, sa3, sa4.
// Each layer dict flattens as W, b, be, g.
// ---------------------------------------------------------------------------
enum { IN_COORD = 0, FP1 = 1, FP2 = 13, FP3 = 21, FP4 = 29,
       HEAD_W = 37, HEAD_B = 38, HEAD_MLP = 39,
       SA1 = 43, SA2 = 67, SA3 = 91, SA4 = 115 };

extern "C" void kernel_launch(void* const* d_in, const int* in_sizes, int n_in,
                              void* d_out, int out_size, void* d_ws, size_t ws_size,
                              hipStream_t stream)
{
  (void)in_sizes; (void)out_size; (void)ws_size;
  if (n_in < 139) return;
  const int B = 16, N0 = 4096;

  char* base = (char*)d_ws;
  size_t off = 0;
  auto alloc = [&](size_t bytes) -> void* {
    void* p = base + off;
    off = (off + bytes + 255) & ~(size_t)255;
    return p;
  };
  auto round64 = [](int x) { return (x + 63) & ~63; };

  // Persistent buffers
  float*  xyz0  = (float*)alloc((size_t)B * N0 * 3 * 4);
  bf16_t* feat0 = (bf16_t*)alloc((size_t)B * N0 * 16 * 2);
  int*    idxb  = (int*)alloc((size_t)B * 1024 * 4);
  float*  nxyz1 = (float*)alloc((size_t)B * 1024 * 3 * 4);
  bf16_t* feat1 = (bf16_t*)alloc((size_t)B * 1024 * 96 * 2);
  float*  nxyz2 = (float*)alloc((size_t)B * 256 * 3 * 4);
  bf16_t* feat2 = (bf16_t*)alloc((size_t)B * 256 * 256 * 2);
  float*  nxyz3 = (float*)alloc((size_t)B * 64 * 3 * 4);
  bf16_t* feat3 = (bf16_t*)alloc((size_t)B * 64 * 512 * 2);
  float*  nxyz4 = (float*)alloc((size_t)B * 16 * 3 * 4);
  bf16_t* feat4 = (bf16_t*)alloc((size_t)B * 16 * 1024 * 2);
  bf16_t* fp4o  = (bf16_t*)alloc((size_t)B * 64 * 256 * 2);
  bf16_t* fp3o  = (bf16_t*)alloc((size_t)B * 256 * 256 * 2);
  bf16_t* fp2o  = (bf16_t*)alloc((size_t)B * 1024 * 128 * 2);
  bf16_t* fp1o  = (bf16_t*)alloc((size_t)B * 4096 * 128 * 2);
  float*  logitsF = (float*)alloc((size_t)B * 4096 * 64 * 4);

  const size_t BIG = (size_t)524288 * 64 * 2;  // max rows x max padded channels
  bf16_t* G0 = (bf16_t*)alloc(BIG);
  bf16_t* G1 = (bf16_t*)alloc(BIG);
  bf16_t* G2 = (bf16_t*)alloc(BIG);

  // GEMM launcher (stages weights each call -> deterministic, no caching)
  auto gemm_ptrs = [&](const float* W, const float* bias, const float* g, const float* be,
                       int cin, int cout, const bf16_t* Xin, int M,
                       bf16_t* Y, float* Yf, bool bnrelu) -> int {
    int Kp = round64(cin), Np = round64(cout);
    bf16_t* Wp = (bf16_t*)alloc((size_t)Np * Kp * sizeof(bf16_t));
    float*  ep = (float*)alloc((size_t)3 * Np * sizeof(float));
    int tot = Np * Kp;
    prep_weights<<<(tot + 255) / 256, 256, 0, stream>>>(W, bias, g, be, Wp, ep, cout, cin, Np, Kp);
    int blocks = (M >> 7) * (Np >> 6);           // 128-row x 64-col stripes
    wmma_gemm_tiled<<<blocks, 256, 0, stream>>>(Xin, Wp, ep, Y, Yf, M, Np, Kp, Np,
                                                bnrelu ? 1 : 0);
    return Np;
  };
  auto gemm_idx = [&](int pidx, int cin, int cout, const bf16_t* Xin, int M,
                      bf16_t* Y, float* Yf, bool bnrelu) {
    return gemm_ptrs((const float*)d_in[pidx], (const float*)d_in[pidx + 1],
                     (const float*)d_in[pidx + 3], (const float*)d_in[pidx + 2],
                     cin, cout, Xin, M, Y, Yf, bnrelu);
  };

  auto sa_branch = [&](const float* xyz_in, int Nin, const bf16_t* feat_in, int Cf, int ldf,
                       const float* nxyz, int S, float radius, int K, const int* d, int pb,
                       bf16_t* feat_out, int ldo, int coff) {
    int Kp = round64(d[0]);
    ball_gather<<<B * S, 64, 0, stream>>>(xyz_in, nxyz, feat_in, Cf, ldf,
                                          G0, Nin, S, K, Kp, radius * radius);
    int M = B * S * K;
    gemm_idx(pb + 0, d[0], d[1], G0, M, G1, nullptr, true);
    gemm_idx(pb + 4, d[1], d[2], G1, M, G2, nullptr, true);
    int ld3 = gemm_idx(pb + 8, d[2], d[3], G2, M, G1, nullptr, true);
    maxpool_kernel<<<B * S, 128, 0, stream>>>(G1, feat_out, K, d[3], ld3, ldo, coff);
  };

  auto sa_stage = [&](const float* xyz_in, int Nin, const bf16_t* feat_in, int Cf, int ldf,
                      int S, float r0, int K0, const int* d0, int pb,
                      float r1, int K1, const int* d1,
                      float* nxyz, bf16_t* feat_out, int ldo) {
    fps_kernel<<<B, 256, 0, stream>>>(xyz_in, Nin, S, idxb, nxyz);
    sa_branch(xyz_in, Nin, feat_in, Cf, ldf, nxyz, S, r0, K0, d0, pb,      feat_out, ldo, 0);
    sa_branch(xyz_in, Nin, feat_in, Cf, ldf, nxyz, S, r1, K1, d1, pb + 12, feat_out, ldo, d0[3]);
  };

  auto fp_stage = [&](const float* xyz1, int N1, const bf16_t* f1, int C1, int ld1,
                      const float* xyz2, int N2, const bf16_t* f2, int C2, int ld2,
                      const int* d, int nl, int pb, bf16_t* outBuf) {
    int Kp = round64(d[0]);
    int M = B * N1;
    fp_gather<<<(M + 255) / 256, 256, 0, stream>>>(xyz1, xyz2, f1, C1, ld1, f2, C2, ld2,
                                                   G0, N1, N2, Kp, M);
    const bf16_t* src = G0; int kc = d[0];
    for (int l = 0; l < nl; ++l) {
      bf16_t* dst = (l == nl - 1) ? outBuf : ((l & 1) ? G2 : G1);
      gemm_idx(pb + 4 * l, kc, d[l + 1], src, M, dst, nullptr, true);
      src = dst; kc = d[l + 1];
    }
  };

  const int sa1d0[4] = {12, 16, 16, 32},     sa1d1[4] = {12, 32, 32, 64};
  const int sa2d0[4] = {99, 64, 64, 128},    sa2d1[4] = {99, 64, 96, 128};
  const int sa3d0[4] = {259, 128, 196, 256}, sa3d1[4] = {259, 128, 196, 256};
  const int sa4d0[4] = {515, 256, 256, 512}, sa4d1[4] = {515, 256, 384, 512};
  const int fp4d[3] = {1536, 256, 256}, fp3d[3] = {512, 256, 256};
  const int fp2d[3] = {352, 256, 128},  fp1d[4] = {137, 128, 128, 128};

  // --- Pipeline ---
  prep_coord<<<(B * N0 + 255) / 256, 256, 0, stream>>>(
      (const float*)d_in[IN_COORD], xyz0, feat0, B, N0, 9, 16);

  sa_stage(xyz0,  4096, feat0, 9,   16,  1024, 0.05f, 16, sa1d0, SA1, 0.1f, 32, sa1d1, nxyz1, feat1, 96);
  sa_stage(nxyz1, 1024, feat1, 96,  96,  256,  0.1f,  16, sa2d0, SA2, 0.2f, 32, sa2d1, nxyz2, feat2, 256);
  sa_stage(nxyz2, 256,  feat2, 256, 256, 64,   0.2f,  16, sa3d0, SA3, 0.4f, 32, sa3d1, nxyz3, feat3, 512);
  sa_stage(nxyz3, 64,   feat3, 512, 512, 16,   0.4f,  16, sa4d0, SA4, 0.8f, 32, sa4d1, nxyz4, feat4, 1024);

  fp_stage(nxyz3, 64,   feat3, 512, 512,  nxyz4, 16,   feat4, 1024, 1024, fp4d, 2, FP4, fp4o);
  fp_stage(nxyz2, 256,  feat2, 256, 256,  nxyz3, 64,   fp4o,  256,  256,  fp3d, 2, FP3, fp3o);
  fp_stage(nxyz1, 1024, feat1, 96,  96,   nxyz2, 256,  fp3o,  256,  256,  fp2d, 2, FP2, fp2o);
  fp_stage(xyz0,  4096, feat0, 9,   16,   nxyz1, 1024, fp2o,  128,  128,  fp1d, 3, FP1, fp1o);

  // Head MLP + logits + log_softmax
  gemm_idx(HEAD_MLP, 128, 128, fp1o, B * 4096, G1, nullptr, true);
  gemm_ptrs((const float*)d_in[HEAD_W], (const float*)d_in[HEAD_B],
            (const float*)d_in[HEAD_B], (const float*)d_in[HEAD_B],
            128, 13, G1, B * 4096, nullptr, logitsF, false);
  logsoftmax_kernel<<<(B * 4096 + 255) / 256, 256, 0, stream>>>(
      logitsF, 64, 13, (float*)d_out, B * 4096);
  newxyz_kernel<<<3, 256, 0, stream>>>(nxyz4, (float*)d_out + (size_t)B * 4096 * 13);
}
